// Decompressor_26792005992938
// MI455X (gfx1250) — matrix operand order repaired and checked by reading, not compiled, not run on hardware
//
#include <hip/hip_runtime.h>
#include <hip/hip_bf16.h>
#include <cstdint>
#include <cstddef>

#define EMBED 2048
#define T_STEPS 2048
#define GATES 8192

typedef __attribute__((ext_vector_type(16))) __bf16 v16bf;
typedef __attribute__((ext_vector_type(8)))  float  v8f;

struct R8 { unsigned int w[8]; };

// Load 32 bytes (16 bf16) from two 16B-aligned chunks into a WMMA fragment.
__device__ __forceinline__ v16bf ld_frag(const unsigned short* p0, const unsigned short* p1) {
    R8 r;
    uint4 a = *reinterpret_cast<const uint4*>(p0);
    uint4 b = *reinterpret_cast<const uint4*>(p1);
    r.w[0]=a.x; r.w[1]=a.y; r.w[2]=a.z; r.w[3]=a.w;
    r.w[4]=b.x; r.w[5]=b.y; r.w[6]=b.z; r.w[7]=b.w;
    return __builtin_bit_cast(v16bf, r);
}

__device__ __forceinline__ unsigned short f2bf(float f) {
    unsigned int u = __float_as_uint(f);
    u += 0x7fffu + ((u >> 16) & 1u);   // round-to-nearest-even
    return (unsigned short)(u >> 16);
}

__device__ __forceinline__ float sigmoidf_(float x) {
    return 1.0f / (1.0f + __expf(-x));
}

// ---------------------------------------------------------------------------
// Prep: bsum = b_ih + b_hh ; h16 double buffer (buf0 rows0-1 = bf16(x),
// everything else zero) ; cbuf = 0
// ---------------------------------------------------------------------------
__global__ void prep_small(const float* __restrict__ x,
                           const float* __restrict__ b_ih,
                           const float* __restrict__ b_hh,
                           float* __restrict__ bsum,
                           unsigned short* __restrict__ h16,
                           float* __restrict__ cbuf) {
    int i = blockIdx.x * blockDim.x + threadIdx.x;
    if (i < GATES) bsum[i] = b_ih[i] + b_hh[i];
    if (i < 2 * 16 * EMBED) {
        int e   = i % EMBED;
        int row = (i / EMBED) & 15;
        int buf = i / (16 * EMBED);
        float v = (buf == 0 && row < 2) ? x[row * EMBED + e] : 0.0f;
        h16[i] = f2bf(v);
    }
    if (i < 2 * EMBED) cbuf[i] = 0.0f;
}

// ---------------------------------------------------------------------------
// Pack W_ih and (W_ih+W_hh) as bf16 in WMMA B-fragment order.
// Chunk holds W[tg*16 + (lane&15)][kc*32 + (lane>>4)*8 + {0..7, 16..23}].
// ---------------------------------------------------------------------------
__global__ void pack_w(const float* __restrict__ Wih, const float* __restrict__ Whh,
                       unsigned short* __restrict__ WihP, unsigned short* __restrict__ WsumP) {
    int idx  = blockIdx.x * blockDim.x + threadIdx.x;     // 0 .. 8192*2048/16-1
    int lane = idx & 31;
    int kc   = (idx >> 5) & 63;
    int tg   = idx >> 11;
    int j = tg * 16 + (lane & 15);
    int k = kc * 32 + (lane >> 4) * 8;
    const float* pI = Wih + (size_t)j * EMBED + k;
    const float* pH = Whh + (size_t)j * EMBED + k;
    size_t o = (size_t)idx * 16;
    #pragma unroll
    for (int v = 0; v < 16; ++v) {
        float a = pI[v < 8 ? v : v + 8], b = pH[v < 8 ? v : v + 8];
        WihP[o + v]  = f2bf(a);
        WsumP[o + v] = f2bf(a + b);
    }
}

__global__ void pack_single(const float* __restrict__ W, unsigned short* __restrict__ WP) {
    int idx  = blockIdx.x * blockDim.x + threadIdx.x;     // 0 .. 2048*2048/16-1
    int lane = idx & 31;
    int kc   = (idx >> 5) & 63;
    int tg   = idx >> 11;
    int j = tg * 16 + (lane & 15);
    int k = kc * 32 + (lane >> 4) * 8;
    const float* p = W + (size_t)j * EMBED + k;
    size_t o = (size_t)idx * 16;
    #pragma unroll
    for (int v = 0; v < 16; ++v) WP[o + v] = f2bf(p[v < 8 ? v : v + 8]);
}

// ---------------------------------------------------------------------------
// One LSTM timestep. Grid: 128 blocks x 128 threads (4 waves).
// Block owns 16 hidden units; wave q computes gate q (i,f,g,o) via bf16 WMMA
// over K=2048 with a depth-4 software pipeline. Gate results meet in LDS,
// then the fp32 cell update runs on lanes 0-15 of wave 0.
// h is double-buffered across steps.
// ---------------------------------------------------------------------------
__global__ void __launch_bounds__(128) lstm_step(
        const unsigned short* __restrict__ Wp,   // packed weights (WihP at t==0, else WsumP)
        const float* __restrict__ bsum,
        const unsigned short* __restrict__ hin,  // [16][EMBED] bf16 (rows 2-15 zero)
        unsigned short* __restrict__ hout,       // [16][EMBED] bf16
        float* __restrict__ cbuf,                // [2][EMBED]
        unsigned short* __restrict__ seq,        // [T][2][EMBED] bf16
        int t) {
    __shared__ float sg[4][2][16];               // [gate][batch][unit]

    const int lane = threadIdx.x & 31;
    const int q    = threadIdx.x >> 5;           // gate: 0=i 1=f 2=g 3=o
    const int tile = blockIdx.x;                 // 0..127 -> units e = tile*16 .. +15
    const int m    = lane & 15;
    const int half = lane >> 4;

    const unsigned short* hrow = hin + (size_t)m * EMBED + half * 8;
    const unsigned short* bw   = Wp + ((size_t)(q * 128 + tile) * 64 * 512) + (size_t)lane * 16;

    v8f acc = {};

    // depth-4 software pipeline over 64 K-chunks
    v16bf a0 = ld_frag(hrow +   0, hrow +   0 + 16);
    v16bf w0 = ld_frag(bw   +   0, bw   +   0 + 8);
    v16bf a1 = ld_frag(hrow +  32, hrow +  32 + 16);
    v16bf w1 = ld_frag(bw   + 512, bw   + 512 + 8);
    v16bf a2 = ld_frag(hrow +  64, hrow +  64 + 16);
    v16bf w2 = ld_frag(bw   + 1024, bw  + 1024 + 8);
    v16bf a3 = ld_frag(hrow +  96, hrow +  96 + 16);
    v16bf w3 = ld_frag(bw   + 1536, bw  + 1536 + 8);

    #pragma unroll 4
    for (int kc = 0; kc < 60; ++kc) {
        const int kn = kc + 4;
        v16bf an = ld_frag(hrow + kn * 32,  hrow + kn * 32 + 16);
        v16bf wn = ld_frag(bw   + kn * 512, bw   + kn * 512 + 8);
        acc = __builtin_amdgcn_wmma_f32_16x16x32_bf16(false, a0, false, w0, (short)0, acc, false, false);
        a0 = a1; w0 = w1;
        a1 = a2; w1 = w2;
        a2 = a3; w2 = w3;
        a3 = an; w3 = wn;
    }
    acc = __builtin_amdgcn_wmma_f32_16x16x32_bf16(false, a0, false, w0, (short)0, acc, false, false);
    acc = __builtin_amdgcn_wmma_f32_16x16x32_bf16(false, a1, false, w1, (short)0, acc, false, false);
    acc = __builtin_amdgcn_wmma_f32_16x16x32_bf16(false, a2, false, w2, (short)0, acc, false, false);
    acc = __builtin_amdgcn_wmma_f32_16x16x32_bf16(false, a3, false, w3, (short)0, acc, false, false);

    // D layout: lanes 0-15 hold N=lane, VGPR p = batch row p. Rows 0,1 are real.
    if (lane < 16) {
        sg[q][0][lane] = acc[0];
        sg[q][1][lane] = acc[1];
    }
    __syncthreads();

    if (threadIdx.x < 16) {
        const int e = tile * 16 + threadIdx.x;
        #pragma unroll
        for (int b = 0; b < 2; ++b) {
            float gi = sg[0][b][threadIdx.x] + bsum[e];
            float gf = sg[1][b][threadIdx.x] + bsum[EMBED + e];
            float gg = sg[2][b][threadIdx.x] + bsum[2 * EMBED + e];
            float go = sg[3][b][threadIdx.x] + bsum[3 * EMBED + e];
            float cn = sigmoidf_(gf) * cbuf[b * EMBED + e] + sigmoidf_(gi) * tanhf(gg);
            float hn = sigmoidf_(go) * tanhf(cn);
            cbuf[b * EMBED + e] = cn;
            unsigned short hb = f2bf(hn);
            hout[b * EMBED + e] = hb;
            seq[((size_t)t * 2 + b) * EMBED + e] = hb;
        }
    }
}

// ---------------------------------------------------------------------------
// out[b,t,:] = seq[t,b,:] @ W_out^T + b_out   (M=4096, N=2048, K=2048)
// One wave per 16x16 output tile, bf16 WMMA, f32 accumulate, depth-4 pipeline.
// ---------------------------------------------------------------------------
__global__ void __launch_bounds__(128) out_gemm(
        const unsigned short* __restrict__ seq,   // [4096][2048] bf16, row = t*2+b
        const unsigned short* __restrict__ WoP,   // packed W_out bf16
        const float* __restrict__ b_out,
        float* __restrict__ out) {
    const int lane = threadIdx.x & 31;
    const int wave = threadIdx.x >> 5;
    const int mt = blockIdx.x * 4 + wave;   // 0..255
    const int nt = blockIdx.y;              // 0..127
    const int m    = lane & 15;
    const int half = lane >> 4;

    const unsigned short* arow = seq + (size_t)(mt * 16 + m) * EMBED + half * 8;
    const unsigned short* bp   = WoP + ((size_t)nt * 64 * 512) + (size_t)lane * 16;

    v8f acc = {};

    v16bf a0 = ld_frag(arow +   0, arow +   0 + 16);
    v16bf w0 = ld_frag(bp   +   0, bp   +   0 + 8);
    v16bf a1 = ld_frag(arow +  32, arow +  32 + 16);
    v16bf w1 = ld_frag(bp   + 512, bp   + 512 + 8);
    v16bf a2 = ld_frag(arow +  64, arow +  64 + 16);
    v16bf w2 = ld_frag(bp   + 1024, bp  + 1024 + 8);
    v16bf a3 = ld_frag(arow +  96, arow +  96 + 16);
    v16bf w3 = ld_frag(bp   + 1536, bp  + 1536 + 8);

    #pragma unroll 4
    for (int kc = 0; kc < 60; ++kc) {
        const int kn = kc + 4;
        v16bf an = ld_frag(arow + kn * 32,  arow + kn * 32 + 16);
        v16bf wn = ld_frag(bp   + kn * 512, bp   + kn * 512 + 8);
        acc = __builtin_amdgcn_wmma_f32_16x16x32_bf16(false, a0, false, w0, (short)0, acc, false, false);
        a0 = a1; w0 = w1;
        a1 = a2; w1 = w2;
        a2 = a3; w2 = w3;
        a3 = an; w3 = wn;
    }
    acc = __builtin_amdgcn_wmma_f32_16x16x32_bf16(false, a0, false, w0, (short)0, acc, false, false);
    acc = __builtin_amdgcn_wmma_f32_16x16x32_bf16(false, a1, false, w1, (short)0, acc, false, false);
    acc = __builtin_amdgcn_wmma_f32_16x16x32_bf16(false, a2, false, w2, (short)0, acc, false, false);
    acc = __builtin_amdgcn_wmma_f32_16x16x32_bf16(false, a3, false, w3, (short)0, acc, false, false);

    const int col = nt * 16 + m;
    const float bias = b_out[col];
    #pragma unroll
    for (int p = 0; p < 8; ++p) {
        int r  = mt * 16 + half * 8 + p;   // global row = t*2+b
        int tt = r >> 1, bb = r & 1;
        out[((size_t)bb * T_STEPS + tt) * EMBED + col] = acc[p] + bias;
    }
}

// ---------------------------------------------------------------------------
extern "C" void kernel_launch(void* const* d_in, const int* in_sizes, int n_in,
                              void* d_out, int out_size, void* d_ws, size_t ws_size,
                              hipStream_t stream) {
    const float* x     = (const float*)d_in[0];
    const float* W_ih  = (const float*)d_in[1];
    const float* W_hh  = (const float*)d_in[2];
    const float* b_ih  = (const float*)d_in[3];
    const float* b_hh  = (const float*)d_in[4];
    const float* W_out = (const float*)d_in[5];
    const float* b_out = (const float*)d_in[6];
    float* out = (float*)d_out;

    char* ws = (char*)d_ws;
    unsigned short* WihP  = (unsigned short*)(ws);                 // 33,554,432 B
    unsigned short* WsumP = (unsigned short*)(ws + 33554432);      // 33,554,432 B
    unsigned short* WoutP = (unsigned short*)(ws + 67108864);      //  8,388,608 B
    unsigned short* seqB  = (unsigned short*)(ws + 75497472);      // 16,777,216 B
    float*          bsum  = (float*)         (ws + 92274688);      //     32,768 B
    unsigned short* h16   = (unsigned short*)(ws + 92307456);      //    131,072 B (2 bufs)
    float*          cbuf  = (float*)         (ws + 92438528);      //     16,384 B
    // total ~92.5 MB of d_ws

    prep_small<<<256, 256, 0, stream>>>(x, b_ih, b_hh, bsum, h16, cbuf);
    pack_w<<<4096, 256, 0, stream>>>(W_ih, W_hh, WihP, WsumP);
    pack_single<<<1024, 256, 0, stream>>>(W_out, WoutP);

    for (int t = 0; t < T_STEPS; ++t) {
        const unsigned short* Wp  = (t == 0) ? WihP : WsumP;
        const unsigned short* hin = h16 + (size_t)(t & 1) * 16 * EMBED;
        unsigned short*      hout = h16 + (size_t)((t + 1) & 1) * 16 * EMBED;
        lstm_step<<<128, 128, 0, stream>>>(Wp, bsum, hin, hout, cbuf, seqB, t);
    }

    out_gemm<<<dim3(64, 128), 128, 0, stream>>>(seqB, WoutP, b_out, out);
}
